// GraphCritic_64768106824369
// MI455X (gfx1250) — matrix operand order repaired and checked by reading, not compiled, not run on hardware
//
#include <hip/hip_runtime.h>
#include <hip/hip_bf16.h>
#include <math.h>

typedef __attribute__((ext_vector_type(16))) _Float16 v16h;
typedef __attribute__((ext_vector_type(8)))  float    v8f;

#define NN 100000
#define FF 128
#define HH 64
#define EE 1600000

// ---------------- degree / norm ----------------
__global__ void k_init_deg(float* deg, int n) {
    int i = blockIdx.x * blockDim.x + threadIdx.x;
    if (i < n) deg[i] = 1.0f;                       // self-loop contributes 1
}

__global__ void k_deg_accum(const int* __restrict__ dst, float* deg, int e) {
    int i = blockIdx.x * blockDim.x + threadIdx.x;
    if (i < e) atomicAdd(&deg[dst[i]], 1.0f);
}

__global__ void k_deg_rsqrt(float* deg, int n) {
    int i = blockIdx.x * blockDim.x + threadIdx.x;
    if (i < n) {
        float d = deg[i];
        deg[i] = (d > 0.0f) ? rsqrtf(d) : 0.0f;     // becomes dinv
    }
}

// ---------------- GEMM: xw = x @ conv_W^T via v_wmma_f32_16x16x32_f16 ----------------
// Block = 256 threads = 8 waves. blockIdx.x = 16-row tile; wave w = 16-col tile.
// A fragment (16x32 f16): lane l (group g=l/16, r=l%16) holds row r,
//   elems 0..7 -> K = kb + g*8 + i ; elems 8..15 -> K = kb + 16 + g*8 + (i-8)
// B fragment (32x16 f16): lane l holds column n = l%16,
//   elems 0..7 -> K = kb + g*8 + i ; elems 8..15 -> K = kb + 16 + g*8 + (i-8)  (B[k][n] = W[n*F + k])
// D (16x16 f32, 8 vgprs): lane l holds column n=l%16, vgpr v -> row m = v + 8*g
__global__ void k_gemm_xw(const float* __restrict__ x, const float* __restrict__ W,
                          float* __restrict__ xw) {
    const int wave = threadIdx.x >> 5;         // 0..7 -> column tile
    const int lane = threadIdx.x & 31;
    const int g    = lane >> 4;
    const int lr   = lane & 15;
    const int row  = blockIdx.x * 16 + lr;     // A row for this lane
    const int col  = wave * 16 + lr;           // B/D column for this lane

    v8f acc = {};
#pragma unroll
    for (int ks = 0; ks < 4; ++ks) {
        const int kb = ks * 32;
        v16h a;
        const float* xa0 = x + (size_t)row * FF + kb + g * 8;
        const float* xa1 = xa0 + 16;
#pragma unroll
        for (int i = 0; i < 8; ++i) {
            a[i]     = (_Float16)xa0[i];
            a[i + 8] = (_Float16)xa1[i];
        }
        v16h b;
        const float* wb = W + (size_t)col * FF + kb + g * 8;
#pragma unroll
        for (int i = 0; i < 16; ++i) b[i] = (_Float16)wb[i + (i >= 8 ? 8 : 0)];

        acc = __builtin_amdgcn_wmma_f32_16x16x32_f16(
            /*neg_a=*/false, a, /*neg_b=*/false, b,
            /*c_mod=*/(short)0, acc, /*reuse_a=*/false, /*reuse_b=*/false);
    }
#pragma unroll
    for (int v = 0; v < 8; ++v) {
        int m = v + 8 * g;
        xw[(size_t)(blockIdx.x * 16 + m) * FF + col] = acc[v];
    }
}

// ---------------- h init (vectorized): h = xw*dinv^2 + bias ----------------
__global__ void k_h_init(const float4* __restrict__ xw4, const float* __restrict__ dinv,
                         const float* __restrict__ bias, float4* __restrict__ h4) {
    int i = blockIdx.x * blockDim.x + threadIdx.x;   // 0 .. N*32-1 float4s
    if (i < NN * (FF / 4)) {
        int r  = i >> 5;           // node
        int c4 = (i & 31) << 2;    // feature base
        float dv = dinv[r];
        float s  = dv * dv;
        float4 v = xw4[i];
        float4 o;
        o.x = v.x * s + bias[c4 + 0];
        o.y = v.y * s + bias[c4 + 1];
        o.z = v.z * s + bias[c4 + 2];
        o.w = v.w * s + bias[c4 + 3];
        h4[i] = o;
    }
}

// ---- edge scatter (vectorized): one thread per (edge, float4 chunk) ----
// b128 coalesced gather of xw[src], 4 f32 atomics into h[dst].
__global__ void k_scatter(const int* __restrict__ src, const int* __restrict__ dst,
                          const float* __restrict__ dinv,
                          const float4* __restrict__ xw4,
                          float* __restrict__ h) {
    int gid = blockIdx.x * blockDim.x + threadIdx.x;  // E*32 = 51.2M < 2^31
    int e = gid >> 5;          // edge
    int q = gid & 31;          // float4 chunk within the 128-wide row
    int s = src[e], d = dst[e];
    float w = dinv[s] * dinv[d];
    float4 v = xw4[(size_t)s * (FF / 4) + q];
    float* hp = h + (size_t)d * FF + (q << 2);
    atomicAdd(hp + 0, v.x * w);
    atomicAdd(hp + 1, v.y * w);
    atomicAdd(hp + 2, v.z * w);
    atomicAdd(hp + 3, v.w * w);
}

// ---------------- per-column median via 8-bit MSB radix-select ----------------
__global__ void k_median(const float* __restrict__ h, float* __restrict__ med) {
    const int col = blockIdx.x;
    __shared__ unsigned hist[256];
    __shared__ unsigned sp, sk;
    const int t = threadIdx.x;
    if (t == 0) { sp = 0u; sk = (NN - 1) / 2; }      // k = 49999 ('lower' median)
    __syncthreads();

    for (int pass = 0; pass < 4; ++pass) {
        for (int b = t; b < 256; b += blockDim.x) hist[b] = 0u;
        __syncthreads();
        unsigned prefix = sp;
        int shift = 24 - pass * 8;
        for (int i = t; i < NN; i += blockDim.x) {
            unsigned u = __float_as_uint(h[(size_t)i * FF + col]);
            u = (u & 0x80000000u) ? ~u : (u | 0x80000000u);   // order-preserving map
            if (pass == 0 || (u >> (shift + 8)) == prefix)
                atomicAdd(&hist[(u >> shift) & 0xFFu], 1u);
        }
        __syncthreads();
        if (t == 0) {
            unsigned cum = 0, k = sk, bin = 0;
            for (int b = 0; b < 256; ++b) {
                unsigned c = hist[b];
                if (cum + c > k) { bin = (unsigned)b; break; }
                cum += c;
            }
            sk = k - cum;
            sp = (sp << 8) | bin;
        }
        __syncthreads();
    }
    if (t == 0) {
        unsigned u = sp;
        unsigned bits = (u & 0x80000000u) ? (u ^ 0x80000000u) : ~u;  // inverse map
        med[col] = __uint_as_float(bits);
    }
}

// ---------------- tiny tanh MLP on the 1x128 median vector ----------------
__global__ void k_mlp(const float* __restrict__ med,
                      const float* __restrict__ w1, const float* __restrict__ b1,
                      const float* __restrict__ w2, const float* __restrict__ b2,
                      const float* __restrict__ w3, const float* __restrict__ b3,
                      float* __restrict__ out) {
    __shared__ float sm[FF], a1[HH], a2[HH];
    const int t = threadIdx.x;
    if (t < FF) sm[t] = med[t];
    __syncthreads();
    if (t < HH) {
        float s = b1[t];
        for (int k = 0; k < FF; ++k) s += sm[k] * w1[t * FF + k];
        a1[t] = tanhf(s);
    }
    __syncthreads();
    if (t < HH) {
        float s = b2[t];
        for (int k = 0; k < HH; ++k) s += a1[k] * w2[t * HH + k];
        a2[t] = tanhf(s);
    }
    __syncthreads();
    if (t == 0) {
        float s = b3[0];
        for (int k = 0; k < HH; ++k) s += a2[k] * w3[k];
        out[0] = s;
    }
}

extern "C" void kernel_launch(void* const* d_in, const int* in_sizes, int n_in,
                              void* d_out, int out_size, void* d_ws, size_t ws_size,
                              hipStream_t stream) {
    const float* x      = (const float*)d_in[0];
    const int*   ei     = (const int*)  d_in[1];   // [2, E] flat
    const float* conv_W = (const float*)d_in[2];
    const float* conv_b = (const float*)d_in[3];
    const float* w1     = (const float*)d_in[4];
    const float* b1     = (const float*)d_in[5];
    const float* w2     = (const float*)d_in[6];
    const float* b2     = (const float*)d_in[7];
    const float* w3     = (const float*)d_in[8];
    const float* b3     = (const float*)d_in[9];
    float* out = (float*)d_out;

    const int* src = ei;
    const int* dst = ei + EE;

    float* xw   = (float*)d_ws;                  // N*F
    float* h    = xw + (size_t)NN * FF;          // N*F
    float* dinv = h  + (size_t)NN * FF;          // N
    float* med  = dinv + NN;                     // F

    // degree -> dinv
    k_init_deg <<<(NN + 255) / 256, 256, 0, stream>>>(dinv, NN);
    k_deg_accum<<<(EE + 255) / 256, 256, 0, stream>>>(dst, dinv, EE);
    k_deg_rsqrt<<<(NN + 255) / 256, 256, 0, stream>>>(dinv, NN);

    // xw = x @ W^T  (WMMA f16->f32)
    k_gemm_xw<<<NN / 16, 256, 0, stream>>>(x, conv_W, xw);

    // h = xw*dinv^2 + b  then scatter real edges (float4-vectorized)
    k_h_init<<<(NN * (FF / 4) + 255) / 256, 256, 0, stream>>>(
        (const float4*)xw, dinv, conv_b, (float4*)h);
    k_scatter<<<(EE * (FF / 4)) / 256, 256, 0, stream>>>(
        src, dst, dinv, (const float4*)xw, h);

    // per-column median, then MLP head
    k_median<<<FF, 256, 0, stream>>>(h, med);
    k_mlp<<<1, 128, 0, stream>>>(med, w1, b1, w2, b2, w3, b3, out);
}
